// Transformer_71287867179090
// MI455X (gfx1250) — matrix operand order repaired and checked by reading, not compiled, not run on hardware
//
#include <hip/hip_runtime.h>
#include <math.h>

#define B_   2
#define S_   1024
#define D_   1024
#define H_   16
#define HD_  64
#define V_   32000
#define L_   2
#define DFF_ 4096
#define ROWS (B_ * S_)   // 2048

typedef __attribute__((ext_vector_type(16))) __bf16        v16bf;
typedef __attribute__((ext_vector_type(8)))  float         v8f;
typedef __attribute__((ext_vector_type(8)))  unsigned int  v8u;

// ---------------------------------------------------------------------------
// CDNA5 async global->LDS copy (ASYNCcnt path), guarded so we compile on
// toolchains that do not declare the builtin. Probe round 2 showed the
// builtin's params are pointers to 16B int vectors (global / LDS).
// ---------------------------------------------------------------------------
#if defined(__has_builtin)
#  if __has_builtin(__builtin_amdgcn_global_load_async_to_lds_b128)
#    define ASYNC_LDS 1
#  endif
#  if __has_builtin(__builtin_amdgcn_s_wait_asynccnt)
#    define HAVE_WAIT_ASYNC 1
#  endif
#endif
#ifndef ASYNC_LDS
#  define ASYNC_LDS 0
#endif

__device__ __forceinline__ void async_wait_all() {
#if defined(HAVE_WAIT_ASYNC)
    __builtin_amdgcn_s_wait_asynccnt(0);
#else
    asm volatile("s_wait_asynccnt 0" ::: "memory");
#endif
}

#if ASYNC_LDS
typedef int v4i_vs __attribute__((__vector_size__(4 * sizeof(int))));
__device__ __forceinline__ void async_copy16(const void* g, void* l) {
    // global_load_async_to_lds_b128: per-lane 16B global -> LDS
    __builtin_amdgcn_global_load_async_to_lds_b128(
        (__attribute__((address_space(1))) v4i_vs*)g,
        (__attribute__((address_space(3))) v4i_vs*)l,
        0, 0);
}
#endif

// ---------------------------------------------------------------------------
// helpers
// ---------------------------------------------------------------------------
__device__ __forceinline__ v8f wmma_bf16(v8u a_, v8u b_, v8f c) {
    v16bf a = __builtin_bit_cast(v16bf, a_);
    v16bf b = __builtin_bit_cast(v16bf, b_);
    // D = A(16x32) * B(32x16) + C, f32 accumulate
    return __builtin_amdgcn_wmma_f32_16x16x32_bf16(
        false, a, false, b, (short)0, c, false, false);
}

__device__ __forceinline__ unsigned short bf_bits(__bf16 h) {
    union { __bf16 h; unsigned short u; } t; t.h = h; return t.u;
}

// ---------------------------------------------------------------------------
// fp32 -> bf16 conversion (weights / activations)
// ---------------------------------------------------------------------------
__global__ void k_f32_to_bf16(const float* __restrict__ src,
                              __bf16* __restrict__ dst, int n) {
    int i = blockIdx.x * blockDim.x + threadIdx.x;
    int stride = gridDim.x * blockDim.x;
    for (; i < n; i += stride) dst[i] = (__bf16)src[i];
}

// ---------------------------------------------------------------------------
// x = embedding[tok] * sqrt(D) + sinusoidal PE
// ---------------------------------------------------------------------------
__global__ void k_embed(const int* __restrict__ tok,
                        const float* __restrict__ emb,
                        float* __restrict__ x) {
    const int row = blockIdx.x;          // b*S + s
    const int s   = row % S_;
    const int t   = tok[row];
    const float* e = emb + (size_t)t * D_;
    float* xr = x + (size_t)row * D_;
    const float c0 = -9.210340371976184f / (float)D_;  // -ln(10000)/D
    for (int c = threadIdx.x; c < D_; c += blockDim.x) {
        int odd = c & 1;
        float arg = (float)s * expf((float)(c - odd) * c0);
        float pe  = odd ? cosf(arg) : sinf(arg);
        xr[c] = e[c] * 32.0f + pe;       // sqrt(1024) = 32
    }
}

// ---------------------------------------------------------------------------
// RMSNorm (eps outside sqrt, per reference) -> bf16
// ---------------------------------------------------------------------------
__global__ void k_rms_bf16(const float* __restrict__ x,
                           const float* __restrict__ alpha,
                           __bf16* __restrict__ out, int d) {
    __shared__ float red[256];
    const int row = blockIdx.x;
    const float* xr = x + (size_t)row * d;
    float ss = 0.f;
    for (int c = threadIdx.x; c < d; c += blockDim.x) { float v = xr[c]; ss += v * v; }
    red[threadIdx.x] = ss; __syncthreads();
    for (int off = 128; off > 0; off >>= 1) {
        if (threadIdx.x < off) red[threadIdx.x] += red[threadIdx.x + off];
        __syncthreads();
    }
    const float inv = 1.0f / (sqrtf(red[0] / (float)d) + 1e-8f);
    __bf16* orow = out + (size_t)row * d;
    for (int c = threadIdx.x; c < d; c += blockDim.x)
        orow[c] = (__bf16)(alpha[c] * xr[c] * inv);
}

// ---------------------------------------------------------------------------
// Generic bf16 GEMM: C[M,N] = A[M,K] x B[K,N] (both bf16 row-major).
// Block tile 64x128, 128 threads (4 waves, 2x2), wave tile 32x64
// (acc[2][4] => 8 WMMAs per 32-wide k-step).
// Double-buffered LDS; A tile staged with async global->LDS copies
// (ASYNCcnt) when available, B tile staged regs->LDS transposed so B
// fragments are dword-pair reads. One barrier per k-step; global loads for
// tile t+1 issued before the WMMA block of tile t; t+2 prefetched.
// Requires M%64==0, N%128==0, K%32==0 (true for all call sites).
// ---------------------------------------------------------------------------
template<bool BIAS, bool SILU, bool RESID, bool OUTBF16>
__global__ __launch_bounds__(128)
void k_gemm_bf16(const __bf16* __restrict__ A, const __bf16* __restrict__ Bm,
                 const float* __restrict__ bias, const float* __restrict__ resid,
                 void* __restrict__ outv, int M, int N, int K) {
    // A: 64 rows x 32 k, 16 u32 payload + 4 pad per row (80B stride, 16B-aligned
    //    chunks for b128 async stores, odd bank spread)
    __shared__ __align__(16) unsigned int sA[2][64 * 20];
    // B transposed: 128 cols x 32 k, 16 u32 payload + 1 pad per col
    __shared__ __align__(16) unsigned int sB[2][128 * 17];

    const int tid  = threadIdx.x;
    const int lane = tid & 31;
    const int wave = tid >> 5;       // 0..3
    const int wm   = wave & 1;       // 2 M sub-tiles of 32 rows
    const int wn   = wave >> 1;      // 2 N sub-tiles of 64 cols
    const int hf   = lane >> 4;
    const int l16  = lane & 15;
    const int m0   = blockIdx.y * 64;
    const int n0   = blockIdx.x * 128;

    // ---- staging helpers -------------------------------------------------
    auto issueA = [&](int kb, int buf) {   // 64x32 bf16, 16B per chunk, 2/thread
#pragma unroll
        for (int j = 0; j < 2; ++j) {
            const int c = tid + j * 128;
            const int row = c >> 2, q = c & 3;
            const __bf16* g = A + (size_t)(m0 + row) * K + kb + q * 8;
#if ASYNC_LDS
            char* l = (char*)&sA[buf][0] + row * 80 + q * 16;
            async_copy16(g, l);
#else
            uint4 v = *(const uint4*)g;
            const int base = row * 20 + q * 4;
            sA[buf][base + 0] = v.x; sA[buf][base + 1] = v.y;
            sA[buf][base + 2] = v.z; sA[buf][base + 3] = v.w;
#endif
        }
    };
    auto loadB = [&](int kb, uint4* r) {   // 32x128 bf16, 8 bf16 per chunk, 4/thread
#pragma unroll
        for (int j = 0; j < 4; ++j) {
            const int c = tid + j * 128;
            const int krow = c >> 4, cg = c & 15;
            r[j] = *(const uint4*)(Bm + (size_t)(kb + krow) * N + n0 + cg * 8);
        }
    };
    auto storeB = [&](int buf, const uint4* r) {
        __bf16* sBh = reinterpret_cast<__bf16*>(&sB[buf][0]);
#pragma unroll
        for (int j = 0; j < 4; ++j) {
            const int c = tid + j * 128;
            const int krow = c >> 4, cg = c & 15;
            union { uint4 u; __bf16 h[8]; } t; t.u = r[j];
#pragma unroll
            for (int e = 0; e < 8; ++e)
                sBh[(size_t)(cg * 8 + e) * 34 + krow] = t.h[e];
        }
    };
    auto prefetch_t2 = [&](int kb) {       // hint t+2 tiles toward the caches
        __builtin_prefetch(A  + (size_t)(m0 + (tid >> 2)) * K + kb + (tid & 3) * 8, 0, 1);
        __builtin_prefetch(Bm + (size_t)(kb + (tid >> 4)) * N + n0 + (tid & 15) * 8, 0, 1);
    };

    v8f acc[2][4];
#pragma unroll
    for (int mi = 0; mi < 2; ++mi)
#pragma unroll
        for (int ni = 0; ni < 4; ++ni)
#pragma unroll
            for (int e = 0; e < 8; ++e) acc[mi][ni][e] = 0.f;

    // ---- prologue: stage tile 0 -----------------------------------------
    {
        uint4 rb[4];
        issueA(0, 0);
        loadB(0, rb);
        storeB(0, rb);
        async_wait_all();
        __syncthreads();
    }

    int p = 0;
    for (int kb = 0; kb < K; kb += 32) {
        const bool more = (kb + 32) < K;

        // fragments of tile t from LDS buffer p
        v8u afr[2], bfr[4];
#pragma unroll
        for (int mi = 0; mi < 2; ++mi) {
            const int r = wm * 32 + mi * 16 + l16;
#pragma unroll
            for (int i = 0; i < 8; ++i) {
                const int kk = (i < 4) ? (2 * i + 8 * hf) : (16 + 2 * (i - 4) + 8 * hf);
                afr[mi][i] = sA[p][r * 20 + (kk >> 1)];
            }
        }
#pragma unroll
        for (int ni = 0; ni < 4; ++ni) {
            const int cc = wn * 64 + ni * 16 + l16;
#pragma unroll
            for (int i = 0; i < 8; ++i) {
                const int kk = (i < 4) ? (2 * i + 8 * hf) : (16 + 2 * (i - 4) + 8 * hf);
                bfr[ni][i] = sB[p][cc * 17 + (kk >> 1)];
            }
        }

        // issue tile t+1 traffic before the math
        uint4 rb[4];
        if (more) {
            issueA(kb + 32, p ^ 1);
            loadB(kb + 32, rb);
            if (kb + 64 < K) prefetch_t2(kb + 64);
        }

#pragma unroll
        for (int mi = 0; mi < 2; ++mi)
#pragma unroll
            for (int ni = 0; ni < 4; ++ni)
                acc[mi][ni] = wmma_bf16(afr[mi], bfr[ni], acc[mi][ni]);

        if (more) {
            storeB(p ^ 1, rb);
            async_wait_all();
            __syncthreads();
            p ^= 1;
        }
    }

    // epilogue (C layout: n = lane&15, slot v -> m = v + 8*(lane>>4))
#pragma unroll
    for (int mi = 0; mi < 2; ++mi)
#pragma unroll
        for (int ni = 0; ni < 4; ++ni)
#pragma unroll
            for (int e = 0; e < 8; ++e) {
                const int m = m0 + wm * 32 + mi * 16 + e + 8 * hf;
                const int n = n0 + wn * 64 + ni * 16 + l16;
                float val = acc[mi][ni][e];
                if (BIAS)  val += bias[n];
                if (SILU)  val = val / (1.0f + expf(-val));
                if (RESID) val += resid[(size_t)m * N + n];
                if (OUTBF16) ((__bf16*)outv)[(size_t)m * N + n] = (__bf16)val;
                else         ((float*)outv)[(size_t)m * N + n]  = val;
            }
}

// ---------------------------------------------------------------------------
// Flash attention, causal. One wave per 16-query tile per (b,h).
// Q,K,V,O: bf16 [B,S,D] with head h occupying cols [h*64, h*64+64).
// ---------------------------------------------------------------------------
__global__ __launch_bounds__(32)
void k_attn(const __bf16* __restrict__ Q, const __bf16* __restrict__ Km,
            const __bf16* __restrict__ Vm, __bf16* __restrict__ O) {
    __shared__ unsigned int sP[16 * 17];              // 16x32 bf16 P staging
    __bf16* sPh = reinterpret_cast<__bf16*>(sP);

    const int lane = threadIdx.x;
    const int hf = lane >> 4, l16 = lane & 15;
    const int q0 = blockIdx.x * 16;
    const int h  = blockIdx.y;
    const int b  = blockIdx.z;
    const size_t base = (size_t)b * S_ * D_ + (size_t)h * HD_;
    const float scale = 0.125f;                       // 1/sqrt(64)
    const float NEG = -__builtin_inff();

    // Q fragments: 16 rows x 64 hd as two 16x32 A-fragments (dword loads)
    v8u aq[2];
#pragma unroll
    for (int c = 0; c < 2; ++c)
#pragma unroll
        for (int i = 0; i < 8; ++i) {
            const int kk = ((i < 4) ? (2 * i + 8 * hf) : (16 + 2 * (i - 4) + 8 * hf)) + 32 * c;
            aq[c][i] = *(const unsigned int*)(Q + base + (size_t)(q0 + l16) * D_ + kk);
        }

    float mrun[8], lrun[8];
    v8f oacc[4];
#pragma unroll
    for (int e = 0; e < 8; ++e) { mrun[e] = NEG; lrun[e] = 0.f; }
#pragma unroll
    for (int j = 0; j < 4; ++j)
#pragma unroll
        for (int e = 0; e < 8; ++e) oacc[j][e] = 0.f;

    const int nkt = (q0 + 15) / 32 + 1;               // causal chunk bound
    for (int kt = 0; kt < nkt; ++kt) {
        const int key0 = kt * 32;
        float p[2][8], rmax[8];
#pragma unroll
        for (int e = 0; e < 8; ++e) rmax[e] = NEG;

#pragma unroll
        for (int s = 0; s < 2; ++s) {                 // two 16-key sub-tiles
            const int kb16 = key0 + 16 * s;
            v8f sc;
#pragma unroll
            for (int e = 0; e < 8; ++e) sc[e] = 0.f;
#pragma unroll
            for (int c = 0; c < 2; ++c) {             // hd contraction in 2x32
                v8u bk;                               // B[k=hd][n=key] = K[key][hd]
#pragma unroll
                for (int i = 0; i < 8; ++i) {
                    const int kk = ((i < 4) ? (2 * i + 8 * hf) : (16 + 2 * (i - 4) + 8 * hf)) + 32 * c;
                    bk[i] = *(const unsigned int*)(Km + base + (size_t)(kb16 + l16) * D_ + kk);
                }
                sc = wmma_bf16(aq[c], bk, sc);
            }
#pragma unroll
            for (int e = 0; e < 8; ++e) {             // scale + causal mask
                const int qrow = q0 + e + 8 * hf;
                const int key  = kb16 + l16;
                float val = sc[e] * scale;
                if (key > qrow) val = NEG;
                p[s][e] = val;
                rmax[e] = fmaxf(rmax[e], val);
            }
        }

        // online softmax per row (rows live in half-wave lane groups)
#pragma unroll
        for (int e = 0; e < 8; ++e) {
            float r = rmax[e];
            r = fmaxf(r, __shfl_xor(r, 1));
            r = fmaxf(r, __shfl_xor(r, 2));
            r = fmaxf(r, __shfl_xor(r, 4));
            r = fmaxf(r, __shfl_xor(r, 8));
            const float newm  = fmaxf(mrun[e], r);
            const float alpha = expf(mrun[e] - newm);
            mrun[e] = newm;
            float rs = 0.f;
#pragma unroll
            for (int s = 0; s < 2; ++s) {
                const float ex = expf(p[s][e] - newm);
                p[s][e] = ex;
                rs += ex;
            }
            rs += __shfl_xor(rs, 1);
            rs += __shfl_xor(rs, 2);
            rs += __shfl_xor(rs, 4);
            rs += __shfl_xor(rs, 8);
            lrun[e] = lrun[e] * alpha + rs;
#pragma unroll
            for (int j = 0; j < 4; ++j) oacc[j][e] *= alpha;
        }

        // C-layout P -> A-layout bf16 via LDS bounce
#pragma unroll
        for (int s = 0; s < 2; ++s)
#pragma unroll
            for (int e = 0; e < 8; ++e)
                sPh[(size_t)(e + 8 * hf) * 34 + 16 * s + l16] = (__bf16)p[s][e];
        asm volatile("s_wait_dscnt 0" ::: "memory");

        v8u ap;
#pragma unroll
        for (int i = 0; i < 8; ++i) {
            const int kk = (i < 4) ? (2 * i + 8 * hf) : (16 + 2 * (i - 4) + 8 * hf);
            ap[i] = sP[l16 * 17 + (kk >> 1)];
        }
#pragma unroll
        for (int j = 0; j < 4; ++j) {                 // O += P * V (4 hd tiles)
            v8u bv;                                   // B[k=key][n=hd] = V[key0+k][hd]
#pragma unroll
            for (int i = 0; i < 8; ++i) {
                const int kk = (i < 4) ? (2 * i + 8 * hf) : (16 + 2 * (i - 4) + 8 * hf);
                unsigned int lo = bf_bits(Vm[base + (size_t)(key0 + kk)     * D_ + 16 * j + l16]);
                unsigned int hi = bf_bits(Vm[base + (size_t)(key0 + kk + 1) * D_ + 16 * j + l16]);
                bv[i] = lo | (hi << 16);
            }
            oacc[j] = wmma_bf16(ap, bv, oacc[j]);
        }
    }

    // normalize + store bf16
#pragma unroll
    for (int j = 0; j < 4; ++j)
#pragma unroll
        for (int e = 0; e < 8; ++e) {
            const float o = oacc[j][e] / lrun[e];
            O[base + (size_t)(q0 + e + 8 * hf) * D_ + 16 * j + l16] = (__bf16)o;
        }
}

// ---------------------------------------------------------------------------
// in-place log_softmax over rows of [ROWS, V]
// ---------------------------------------------------------------------------
__global__ void k_logsoftmax(float* __restrict__ out, int n) {
    __shared__ float red[256];
    float* row = out + (size_t)blockIdx.x * n;
    float mx = -__builtin_inff();
    for (int c = threadIdx.x; c < n; c += blockDim.x) mx = fmaxf(mx, row[c]);
    red[threadIdx.x] = mx; __syncthreads();
    for (int off = 128; off > 0; off >>= 1) {
        if (threadIdx.x < off) red[threadIdx.x] = fmaxf(red[threadIdx.x], red[threadIdx.x + off]);
        __syncthreads();
    }
    mx = red[0]; __syncthreads();
    float s = 0.f;
    for (int c = threadIdx.x; c < n; c += blockDim.x) s += expf(row[c] - mx);
    red[threadIdx.x] = s; __syncthreads();
    for (int off = 128; off > 0; off >>= 1) {
        if (threadIdx.x < off) red[threadIdx.x] += red[threadIdx.x + off];
        __syncthreads();
    }
    const float lse = mx + logf(red[0]);
    for (int c = threadIdx.x; c < n; c += blockDim.x) row[c] = row[c] - lse;
}

// ---------------------------------------------------------------------------
// launch
// ---------------------------------------------------------------------------
extern "C" void kernel_launch(void* const* d_in, const int* in_sizes, int n_in,
                              void* d_out, int out_size, void* d_ws, size_t ws_size,
                              hipStream_t stream) {
    (void)in_sizes; (void)n_in; (void)out_size; (void)ws_size;
    const int*   tokens = (const int*)  d_in[0];
    /* d_in[1] = tgt_mask: causal mask applied analytically */
    const float* emb = (const float*)d_in[2];
    const float* Wq  = (const float*)d_in[3];
    const float* Wk  = (const float*)d_in[4];
    const float* Wv  = (const float*)d_in[5];
    const float* Wo  = (const float*)d_in[6];
    const float* a1  = (const float*)d_in[7];
    const float* a3  = (const float*)d_in[8];
    const float* W1  = (const float*)d_in[9];
    const float* b1  = (const float*)d_in[10];
    const float* W2  = (const float*)d_in[11];
    const float* b2  = (const float*)d_in[12];
    const float* afn = (const float*)d_in[13];
    const float* Wp  = (const float*)d_in[14];
    const float* bp  = (const float*)d_in[15];
    float* out = (float*)d_out;

    char* wsp = (char*)d_ws;
    auto carve = [&](size_t bytes) -> char* {
        char* p = wsp; wsp += (bytes + 255) & ~(size_t)255; return p;
    };
    float*  xf = (float*) carve(sizeof(float) * (size_t)ROWS * D_);
    __bf16* nb = (__bf16*)carve(2ull * ROWS * D_);
    __bf16* qb = (__bf16*)carve(2ull * ROWS * D_);
    __bf16* kb = (__bf16*)carve(2ull * ROWS * D_);
    __bf16* vb = (__bf16*)carve(2ull * ROWS * D_);
    __bf16* ob = (__bf16*)carve(2ull * ROWS * D_);
    __bf16* hb = (__bf16*)carve(2ull * ROWS * DFF_);
    __bf16* wb = (__bf16*)carve(2ull * D_ * V_);      // reused weight slot

    k_embed<<<ROWS, 256, 0, stream>>>(tokens, emb, xf);

    for (int l = 0; l < L_; ++l) {
        const size_t dd = (size_t)D_ * D_;
        // pre-norm attention
        k_rms_bf16<<<ROWS, 256, 0, stream>>>(xf, a1 + (size_t)l * D_, nb, D_);
        k_f32_to_bf16<<<1024, 256, 0, stream>>>(Wq + l * dd, wb, D_ * D_);
        k_gemm_bf16<false,false,false,true><<<dim3(D_/128, ROWS/64), 128, 0, stream>>>(
            nb, wb, nullptr, nullptr, qb, ROWS, D_, D_);
        k_f32_to_bf16<<<1024, 256, 0, stream>>>(Wk + l * dd, wb, D_ * D_);
        k_gemm_bf16<false,false,false,true><<<dim3(D_/128, ROWS/64), 128, 0, stream>>>(
            nb, wb, nullptr, nullptr, kb, ROWS, D_, D_);
        k_f32_to_bf16<<<1024, 256, 0, stream>>>(Wv + l * dd, wb, D_ * D_);
        k_gemm_bf16<false,false,false,true><<<dim3(D_/128, ROWS/64), 128, 0, stream>>>(
            nb, wb, nullptr, nullptr, vb, ROWS, D_, D_);

        k_attn<<<dim3(S_/16, H_, B_), 32, 0, stream>>>(qb, kb, vb, ob);

        k_f32_to_bf16<<<1024, 256, 0, stream>>>(Wo + l * dd, wb, D_ * D_);
        k_gemm_bf16<false,false,true,false><<<dim3(D_/128, ROWS/64), 128, 0, stream>>>(
            ob, wb, nullptr, xf, xf, ROWS, D_, D_);

        // pre-norm SiLU FFN
        k_rms_bf16<<<ROWS, 256, 0, stream>>>(xf, a3 + (size_t)l * D_, nb, D_);
        k_f32_to_bf16<<<2048, 256, 0, stream>>>(W1 + (size_t)l * D_ * DFF_, wb, D_ * DFF_);
        k_gemm_bf16<true,true,false,true><<<dim3(DFF_/128, ROWS/64), 128, 0, stream>>>(
            nb, wb, b1 + (size_t)l * DFF_, nullptr, hb, ROWS, DFF_, D_);
        k_f32_to_bf16<<<2048, 256, 0, stream>>>(W2 + (size_t)l * DFF_ * D_, wb, DFF_ * D_);
        k_gemm_bf16<true,false,true,false><<<dim3(D_/128, ROWS/64), 128, 0, stream>>>(
            hb, wb, b2 + (size_t)l * D_, xf, xf, ROWS, D_, DFF_);
    }

    // final norm + vocab projection + log_softmax
    k_rms_bf16<<<ROWS, 256, 0, stream>>>(xf, afn, nb, D_);
    k_f32_to_bf16<<<4096, 256, 0, stream>>>(Wp, wb, D_ * V_);
    k_gemm_bf16<true,false,false,false><<<dim3(V_/128, ROWS/64), 128, 0, stream>>>(
        nb, wb, bp, nullptr, out, ROWS, V_, D_);
    k_logsoftmax<<<ROWS, 256, 0, stream>>>(out, V_);
}
